// TGCN_83313775608099
// MI455X (gfx1250) — matrix operand order repaired and checked by reading, not compiled, not run on hardware
//
#include <hip/hip_runtime.h>
#include <hip/hip_bf16.h>

// ---------------------------------------------------------------------------
// T-GCN forward for MI455X (gfx1250, wave32).  All GEMMs run on the matrix
// pipe via v_wmma_f32_16x16x32_f16 (f16 A/B, f32 accumulate).
// ---------------------------------------------------------------------------

typedef __attribute__((ext_vector_type(16))) _Float16 v16h;
typedef __attribute__((ext_vector_type(8)))  float    v8f;

#define kB   16
#define kT   12
#define kN   1000
#define kH   128
#define kNL  2
#define NP   1024        // kN padded to mult of 64 (M & K of the L-GEMM)
#define KP   160         // (kH+1)=129 padded to mult of 32
#define CPB  144         // per-batch column block (129 -> mult of 16)
#define COLP (kB * CPB)  // 2304 columns of the big graph-conv GEMM
#define BN   (kB * kN)   // 16000 rows (mult of 64)

// ---------------------------------------------------------------------------
// Generic WMMA GEMM:  C[M,Ncols](f32) = A[M,K](f16, row-major, ld=lda)
//                                     x Bt[Ncols,K](f16, row-major, ld=ldb)^T
// M % 64 == 0, Ncols % 128 == 0, K % 32 == 0.  Block = 128 thr (4 waves,
// 2x2 wave grid); each wave owns a 32x64 C footprint = 8 WMMA tiles
// (2 row-frags x 4 col-frags), so each k-step is 12 b128 loads : 8 WMMAs.
// ---------------------------------------------------------------------------
__global__ __launch_bounds__(128) void gemm_wmma_f16(
    const _Float16* __restrict__ A, const _Float16* __restrict__ Bt,
    float* __restrict__ C, int K, int lda, int ldb, int ldc)
{
  const int lane = threadIdx.x & 31;
  const int wave = threadIdx.x >> 5;
  const int wr = wave >> 1, wc = wave & 1;
  const int row0 = blockIdx.y * 64 + wr * 32;
  const int col0 = blockIdx.x * 128 + wc * 64;
  const int m    = lane & 15;
  const int kbA  = (lane & 16) ? 8 : 0;    // A frag: K interleaved 8-blocks
  const int kbB  = (lane & 16) ? 8 : 0;    // B frag: K interleaved 8-blocks

  v8f acc[2][4] = {};
  const _Float16* arow0 = A + (size_t)(row0 + m) * lda;
  const _Float16* arow1 = A + (size_t)(row0 + 16 + m) * lda;

  for (int k0 = 0; k0 < K; k0 += 32) {
    union F16x16 { v16h v; uint4 q[2]; } a[2], b[4];
    a[0].q[0] = *(const uint4*)(arow0 + k0 + kbA);        // K = k0+kb .. +7
    a[0].q[1] = *(const uint4*)(arow0 + k0 + 16 + kbA);   // K = k0+16+kb .. +7
    a[1].q[0] = *(const uint4*)(arow1 + k0 + kbA);
    a[1].q[1] = *(const uint4*)(arow1 + k0 + 16 + kbA);
#pragma unroll
    for (int t = 0; t < 4; ++t) {
      const _Float16* brow = Bt + (size_t)(col0 + t * 16 + m) * ldb + k0 + kbB;
      b[t].q[0] = *(const uint4*)(brow);                  // K = k0+kb .. +7
      b[t].q[1] = *(const uint4*)(brow + 16);             // K = k0+16+kb .. +7
    }
#pragma unroll
    for (int r = 0; r < 2; ++r)
#pragma unroll
      for (int t = 0; t < 4; ++t)
        acc[r][t] = __builtin_amdgcn_wmma_f32_16x16x32_f16(
            false, a[r].v, false, b[t].v, (short)0, acc[r][t], false, false);
  }

  const int mo = (lane & 16) ? 8 : 0;   // C frag: lanes 16-31 hold rows 8..15
#pragma unroll
  for (int r = 0; r < 2; ++r)
#pragma unroll
    for (int t = 0; t < 4; ++t) {
      float* cp = C + (size_t)col0 + t * 16 + m;
#pragma unroll
      for (int q = 0; q < 8; ++q)
        cp[(size_t)(row0 + r * 16 + mo + q) * ldc] = acc[r][t][q];
    }
}

// ---------------------------------------------------------------------------
// Elementwise / pack kernels
// ---------------------------------------------------------------------------
__global__ void k_softmax_rows(const float* __restrict__ adj, float* __restrict__ S)
{
  const int i = blockIdx.x;
  __shared__ float sm[256];
  float mx = -3.4e38f;
  for (int j = threadIdx.x; j < kN; j += 256) mx = fmaxf(mx, adj[i * kN + j]);
  sm[threadIdx.x] = mx; __syncthreads();
  for (int s = 128; s > 0; s >>= 1) {
    if (threadIdx.x < s) sm[threadIdx.x] = fmaxf(sm[threadIdx.x], sm[threadIdx.x + s]);
    __syncthreads();
  }
  mx = sm[0]; __syncthreads();
  float sum = 0.f;
  for (int j = threadIdx.x; j < kN; j += 256) {
    float e = __expf(adj[i * kN + j] - mx);
    S[i * kN + j] = e; sum += e;
  }
  sm[threadIdx.x] = sum; __syncthreads();
  for (int s = 128; s > 0; s >>= 1) {
    if (threadIdx.x < s) sm[threadIdx.x] += sm[threadIdx.x + s];
    __syncthreads();
  }
  const float inv = 1.0f / sm[0];
  for (int j = threadIdx.x; j < kN; j += 256) S[i * kN + j] *= inv;
}

// L[i,j] = (softmax(adj)[j,i] + I) * d_i^-1/2 * d_j^-1/2 ; d == 2 exactly
// (softmax rows sum to 1, +1 from the identity), so the scale is 0.5.
__global__ void k_build_L(const float* __restrict__ S, _Float16* __restrict__ Lh)
{
  int idx = blockIdx.x * 256 + threadIdx.x;
  if (idx >= NP * NP) return;
  int i = idx >> 10, j = idx & (NP - 1);
  float v = 0.f;
  if (i < kN && j < kN) v = 0.5f * (S[j * kN + i] + (i == j ? 1.0f : 0.0f));
  Lh[idx] = (_Float16)v;
}

// wt[o, k] = w[k, o] (k < Kdim), zero-padded to KPad
__global__ void k_wT(const float* __restrict__ w, _Float16* __restrict__ wt,
                     int Kdim, int O, int KPad)
{
  int idx = blockIdx.x * 256 + threadIdx.x;
  if (idx >= O * KPad) return;
  int o = idx / KPad, k = idx - o * KPad;
  wt[idx] = (_Float16)(k < Kdim ? w[k * O + o] : 0.f);
}

__global__ void k_cast16(const float* __restrict__ a, _Float16* __restrict__ o, int n)
{
  int i = blockIdx.x * 256 + threadIdx.x;
  if (i < n) o[i] = (_Float16)a[i];
}

__global__ void k_zero32(float* p, int n)
{
  int i = blockIdx.x * 256 + threadIdx.x;
  if (i < n) p[i] = 0.f;
}
__global__ void k_zero16(_Float16* p, int n)
{
  int i = blockIdx.x * 256 + threadIdx.x;
  if (i < n) p[i] = (_Float16)0.f;
}

// CT[col, n] (f16, [COLP][NP]) = transposed concat([x, (r*)h]) with zero pad.
// col = b*CPB + kc ; kc==0 -> x[b,n] ; 1..128 -> h[b,n,kc-1] (* r-gate if rg).
__global__ void k_pack_concatT(const float* __restrict__ x, int xstride,
                               const float* __restrict__ h,
                               const float* __restrict__ rg,
                               _Float16* __restrict__ CT)
{
  int idx = blockIdx.x * 256 + threadIdx.x;
  if (idx >= COLP * NP) return;
  int col = idx >> 10, n = idx & (NP - 1);
  int b = col / CPB, kc = col - b * CPB;
  float v = 0.f;
  if (n < kN) {
    if (kc == 0) v = x[b * xstride + n];
    else if (kc <= kH) {
      int k = kc - 1;
      float hv = h[((size_t)b * kN + n) * kH + k];
      if (rg) hv *= rg[(size_t)b * (kN * 2 * kH) + (size_t)n * kH + k];
      v = hv;
    }
  }
  CT[idx] = (_Float16)v;
}

// A1[(b*kN+n), k] (f16, [BN][KP]) = AC[n, b*CPB + k] (k < 129), zero pad.
__global__ void k_pack_rows(const float* __restrict__ AC, _Float16* __restrict__ A1)
{
  int idx = blockIdx.x * 256 + threadIdx.x;
  if (idx >= BN * KP) return;
  int r = idx / KP, k = idx - r * KP;
  int b = r / kN, n = r - b * kN;
  float v = (k <= kH) ? AC[(size_t)n * COLP + b * CPB + k] : 0.f;
  A1[idx] = (_Float16)v;
}

__global__ void k_bias_sigmoid(float* __restrict__ G, const float* __restrict__ bias,
                               int cols, int total)
{
  int idx = blockIdx.x * 256 + threadIdx.x;
  if (idx >= total) return;
  int c = idx % cols;
  float v = G[idx] + bias[c];
  G[idx] = 1.0f / (1.0f + __expf(-v));
}

// Per-(b,n) finish: c=tanh(..), h=u*h+(1-u)*c, +residual Linear(1,H),
// LayerNorm, x_out = mean_H(out); optionally emit f16 GRU input row.
__global__ __launch_bounds__(128) void k_cell_finish(
    const float* __restrict__ G,   const float* __restrict__ C2,
    const float* __restrict__ gc2b, float* __restrict__ hbuf,
    const float* __restrict__ xin, int xstride,
    const float* __restrict__ resw, const float* __restrict__ resb,
    const float* __restrict__ lng,  const float* __restrict__ lnb,
    float* __restrict__ xout, _Float16* __restrict__ Xh)
{
  const int bn = blockIdx.x;
  const int b = bn / kN, n = bn - b * kN;
  const int k = threadIdx.x;
  __shared__ float sm[128];

  // torch chunks the FLAT [B, N*2H] tensor: u lives at flat offset N*H.
  float u    = G[(size_t)b * (kN * 2 * kH) + (size_t)kN * kH + (size_t)n * kH + k];
  float c    = tanhf(C2[(size_t)bn * kH + k] + gc2b[k]);
  float hold = hbuf[(size_t)bn * kH + k];
  float hn   = u * hold + (1.0f - u) * c;
  hbuf[(size_t)bn * kH + k] = hn;

  float xv = xin[b * xstride + n];
  float y  = hn + xv * resw[k] + resb[k];

  sm[k] = y; __syncthreads();
  for (int s = 64; s > 0; s >>= 1) { if (k < s) sm[k] += sm[k + s]; __syncthreads(); }
  float mu = sm[0] * (1.0f / kH); __syncthreads();
  float d = y - mu;
  sm[k] = d * d; __syncthreads();
  for (int s = 64; s > 0; s >>= 1) { if (k < s) sm[k] += sm[k + s]; __syncthreads(); }
  float var = sm[0] * (1.0f / kH); __syncthreads();
  float o = d * rsqrtf(var + 1e-5f) * lng[k] + lnb[k];

  if (Xh) Xh[(size_t)bn * kH + k] = (_Float16)o;

  sm[k] = o; __syncthreads();
  for (int s = 64; s > 0; s >>= 1) { if (k < s) sm[k] += sm[k + s]; __syncthreads(); }
  if (k == 0) xout[bn] = sm[0] * (1.0f / kH);
}

__global__ void k_gru_update(const float* __restrict__ GI, const float* __restrict__ GH,
                             const float* __restrict__ bih, const float* __restrict__ bhh,
                             float* __restrict__ hbuf, _Float16* __restrict__ Hh, int total)
{
  int idx = blockIdx.x * 256 + threadIdx.x;
  if (idx >= total) return;
  int row = idx / kH, k = idx - row * kH;
  size_t base = (size_t)row * (3 * kH);
  float ir = GI[base + k] + bih[k];
  float iz = GI[base + kH + k] + bih[kH + k];
  float in = GI[base + 2 * kH + k] + bih[2 * kH + k];
  float hr = GH[base + k] + bhh[k];
  float hz = GH[base + kH + k] + bhh[kH + k];
  float hn = GH[base + 2 * kH + k] + bhh[2 * kH + k];
  float r = 1.0f / (1.0f + __expf(-(ir + hr)));
  float z = 1.0f / (1.0f + __expf(-(iz + hz)));
  float nn = tanhf(in + r * hn);
  float hp = hbuf[idx];
  float h2 = (1.0f - z) * nn + z * hp;
  hbuf[idx] = h2;
  Hh[idx] = (_Float16)h2;
}

__global__ void k_bias_act_pack(const float* __restrict__ in, const float* __restrict__ bias,
                                _Float16* __restrict__ out, int cols, int relu, int total)
{
  int idx = blockIdx.x * 256 + threadIdx.x;
  if (idx >= total) return;
  float v = in[idx] + bias[idx % cols];
  if (relu) v = fmaxf(v, 0.f);
  out[idx] = (_Float16)v;
}

__global__ void k_bias_out(const float* __restrict__ in, const float* __restrict__ bias,
                           float* __restrict__ out, int cols, int total)
{
  int idx = blockIdx.x * 256 + threadIdx.x;
  if (idx >= total) return;
  out[idx] = in[idx] + bias[idx % cols];
}

// ---------------------------------------------------------------------------
// Orchestration
// ---------------------------------------------------------------------------
static inline dim3 gemm_grid(int M, int Ncols) { return dim3(Ncols / 128, M / 64); }

extern "C" void kernel_launch(void* const* d_in, const int* in_sizes, int n_in,
                              void* d_out, int out_size, void* d_ws, size_t ws_size,
                              hipStream_t stream)
{
  const float* inputs  = (const float*)d_in[0];
  const float* adj     = (const float*)d_in[1];
  const float* gc1_w   = (const float*)d_in[2];
  const float* gc1_b   = (const float*)d_in[3];
  const float* gc2_w   = (const float*)d_in[4];
  const float* gc2_b   = (const float*)d_in[5];
  const float* ln_g    = (const float*)d_in[6];
  const float* ln_b    = (const float*)d_in[7];
  const float* res_w   = (const float*)d_in[8];
  const float* res_b   = (const float*)d_in[9];
  const float* gru_wih = (const float*)d_in[10];
  const float* gru_whh = (const float*)d_in[11];
  const float* gru_bih = (const float*)d_in[12];
  const float* gru_bhh = (const float*)d_in[13];
  const float* proj_w  = (const float*)d_in[14];
  const float* proj_b  = (const float*)d_in[15];
  const float* out_w1  = (const float*)d_in[16];
  const float* out_b1  = (const float*)d_in[17];
  const float* out_w2  = (const float*)d_in[18];
  const float* out_b2  = (const float*)d_in[19];

  char* ws = (char*)d_ws;
  size_t off = 0;
  auto alloc = [&](size_t bytes) -> char* {
    char* p = ws + off;
    off += (bytes + 255) & ~(size_t)255;
    return p;
  };

  // Persistent buffers (~52.2 MB)
  _Float16* Lh    = (_Float16*)alloc((size_t)NP * NP * 2);
  _Float16* gc1T  = (_Float16*)alloc((size_t)kNL * 256 * KP * 2);
  _Float16* gc2T  = (_Float16*)alloc((size_t)kNL * 128 * KP * 2);
  _Float16* wihT  = (_Float16*)alloc((size_t)384 * 128 * 2);
  _Float16* whhT  = (_Float16*)alloc((size_t)384 * 128 * 2);
  _Float16* projT = (_Float16*)alloc((size_t)128 * 128 * 2);
  _Float16* mw1   = (_Float16*)alloc((size_t)256 * 128 * 2);
  _Float16* mw2   = (_Float16*)alloc((size_t)128 * 256 * 2);
  float*    xbuf  = (float*)   alloc((size_t)BN * 4);
  _Float16* Xh    = (_Float16*)alloc((size_t)kT * BN * kH * 2);

  // Phase-aliased region (64.3 MB); total ws need ~111 MiB.
  char* R = alloc(64235776);
  // recurrent-phase mapping
  _Float16* CT = (_Float16*)(R + 0);          // 4,718,592  B
  float*    AC = (float*)   (R + 4718592);    // 9,437,184
  _Float16* A1 = (_Float16*)(R + 14155776);   // 5,120,000
  float*    G  = (float*)   (R + 19275776);   // 16,384,000
  float*    C2 = (float*)   (R + 35659776);   // 8,192,000
  float*    hb = (float*)   (R + 43851776);   // 16,384,000 (both layers)
  float*    S  = (float*)   (R + 60235776);   // 4,000,000  (setup only)
  // GRU-phase mapping (recurrent scratch dead by then)
  float*    GI = (float*)   (R + 0);          // 24,576,000
  float*    GH = (float*)   (R + 24576000);   // 24,576,000
  float*    hg = (float*)   (R + 49152000);   // 8,192,000
  _Float16* Hh = (_Float16*)(R + 57344000);   // 4,096,000
  // tail mapping (aliases GI/GH; hg/Hh stay live)
  float*    P    = (float*)   (R + 0);        // 8,192,000
  _Float16* pf16 = (_Float16*)(R + 8192000);  // 4,096,000
  float*    M1   = (float*)   (R + 12288000); // 16,384,000
  _Float16* m1h  = (_Float16*)(R + 28672000); // 8,192,000
  float*    M2   = (float*)   (R + 36864000); // 8,192,000

  // ---- Setup: Laplacian + weight packing -------------------------------
  k_softmax_rows<<<kN, 256, 0, stream>>>(adj, S);
  k_build_L<<<(NP * NP) / 256, 256, 0, stream>>>(S, Lh);
  for (int l = 0; l < kNL; ++l) {
    k_wT<<<(256 * KP + 255) / 256, 256, 0, stream>>>(
        gc1_w + (size_t)l * 129 * 256, gc1T + (size_t)l * 256 * KP, 129, 256, KP);
    k_wT<<<(128 * KP + 255) / 256, 256, 0, stream>>>(
        gc2_w + (size_t)l * 129 * 128, gc2T + (size_t)l * 128 * KP, 129, 128, KP);
  }
  k_cast16<<<(384 * 128 + 255) / 256, 256, 0, stream>>>(gru_wih, wihT, 384 * 128);
  k_cast16<<<(384 * 128 + 255) / 256, 256, 0, stream>>>(gru_whh, whhT, 384 * 128);
  k_cast16<<<(128 * 128 + 255) / 256, 256, 0, stream>>>(proj_w, projT, 128 * 128);
  k_cast16<<<(256 * 128 + 255) / 256, 256, 0, stream>>>(out_w1, mw1, 256 * 128);
  k_cast16<<<(128 * 256 + 255) / 256, 256, 0, stream>>>(out_w2, mw2, 128 * 256);
  k_zero32<<<(kNL * BN * kH + 255) / 256, 256, 0, stream>>>(hb, kNL * BN * kH);

  // ---- Recurrent graph-conv GRU cells ----------------------------------
  const int packCT_g = (COLP * NP) / 256;
  const int packA1_g = (BN * KP + 255) / 256;
  for (int t = 0; t < kT; ++t) {
    for (int l = 0; l < kNL; ++l) {
      const float* xptr = (l == 0) ? (inputs + (size_t)t * kN) : xbuf;
      const int xstride = (l == 0) ? (kT * kN) : kN;
      float* hl = hb + (size_t)l * BN * kH;

      // gate path: sigmoid(gconv(L, [x,h]) @ W1 + b1)
      k_pack_concatT<<<packCT_g, 256, 0, stream>>>(xptr, xstride, hl, nullptr, CT);
      gemm_wmma_f16<<<gemm_grid(NP, COLP), 128, 0, stream>>>(Lh, CT, AC, NP, NP, NP, COLP);
      k_pack_rows<<<packA1_g, 256, 0, stream>>>(AC, A1);
      gemm_wmma_f16<<<gemm_grid(BN, 256), 128, 0, stream>>>(
          A1, gc1T + (size_t)l * 256 * KP, G, KP, KP, KP, 256);
      k_bias_sigmoid<<<(BN * 256) / 256, 256, 0, stream>>>(G, gc1_b + l * 256, 256, BN * 256);

      // candidate path: tanh(gconv(L, [x, r*h]) @ W2 + b2)
      k_pack_concatT<<<packCT_g, 256, 0, stream>>>(xptr, xstride, hl, G, CT);
      gemm_wmma_f16<<<gemm_grid(NP, COLP), 128, 0, stream>>>(Lh, CT, AC, NP, NP, NP, COLP);
      k_pack_rows<<<packA1_g, 256, 0, stream>>>(AC, A1);
      gemm_wmma_f16<<<gemm_grid(BN, 128), 128, 0, stream>>>(
          A1, gc2T + (size_t)l * 128 * KP, C2, KP, KP, KP, 128);

      k_cell_finish<<<BN, 128, 0, stream>>>(
          G, C2, gc2_b + l * 128, hl, xptr, xstride, res_w, res_b,
          ln_g + l * 128, ln_b + l * 128, xbuf,
          (l == kNL - 1) ? (Xh + (size_t)t * BN * kH) : nullptr);
    }
  }

  // ---- GRU tail over time (B*N = 16000 sequences) ----------------------
  k_zero32<<<(BN * kH + 255) / 256, 256, 0, stream>>>(hg, BN * kH);
  k_zero16<<<(BN * kH + 255) / 256, 256, 0, stream>>>(Hh, BN * kH);
  for (int t = 0; t < kT; ++t) {
    gemm_wmma_f16<<<gemm_grid(BN, 384), 128, 0, stream>>>(
        Xh + (size_t)t * BN * kH, wihT, GI, 128, 128, 128, 384);
    gemm_wmma_f16<<<gemm_grid(BN, 384), 128, 0, stream>>>(Hh, whhT, GH, 128, 128, 128, 384);
    k_gru_update<<<(BN * kH) / 256, 256, 0, stream>>>(
        GI, GH, gru_bih, gru_bhh, hg, Hh, BN * kH);
  }

  // ---- Projection + output MLP -----------------------------------------
  gemm_wmma_f16<<<gemm_grid(BN, 128), 128, 0, stream>>>(Hh, projT, P, 128, 128, 128, 128);
  k_bias_act_pack<<<(BN * 128) / 256, 256, 0, stream>>>(P, proj_b, pf16, 128, 0, BN * 128);
  gemm_wmma_f16<<<gemm_grid(BN, 256), 128, 0, stream>>>(pf16, mw1, M1, 128, 128, 128, 256);
  k_bias_act_pack<<<(BN * 256) / 256, 256, 0, stream>>>(M1, out_b1, m1h, 256, 1, BN * 256);
  gemm_wmma_f16<<<gemm_grid(BN, 128), 128, 0, stream>>>(m1h, mw2, M2, 256, 256, 256, 128);
  k_bias_out<<<(BN * 128) / 256, 256, 0, stream>>>(M2, out_b2, (float*)d_out, 128, BN * 128);

  (void)in_sizes; (void)n_in; (void)out_size; (void)ws_size;
}
